// GIN0_2611340116520
// MI455X (gfx1250) — matrix operand order repaired and checked by reading, not compiled, not run on hardware
//
#include <hip/hip_runtime.h>
#include <hip/hip_bf16.h>

#define N_NODES 100000
#define N_EDGES 1600000
#define HDIM    128
#define NLAYERS 4
#define NGRAPH  128
#define NCLASS  10
#define BN_EPS  1e-5f

typedef __attribute__((ext_vector_type(16))) __bf16 v16bf;
typedef __attribute__((ext_vector_type(8)))  float  v8f;

// ---------------- edge scatter-add: agg[dst] += h[src] ----------------
// One edge per wave32: index loads scalarized via readfirstlane, each lane
// moves 4 contiguous floats (coalesced b128 load + 4 f32 atomics).
__global__ __launch_bounds__(256) void gin_scatter(const float* __restrict__ h,
                                                   const int* __restrict__ srcv,
                                                   const int* __restrict__ dstv,
                                                   float* __restrict__ agg) {
  const int wave = threadIdx.x >> 5;
  const int lane = threadIdx.x & 31;
  const int e = blockIdx.x * 8 + wave;               // E % 8 == 0
  const int s = __builtin_amdgcn_readfirstlane(srcv[e]);
  const int d = __builtin_amdgcn_readfirstlane(dstv[e]);
  const float4 v = *(const float4*)(h + (size_t)s * HDIM + lane * 4);
  float* out = agg + (size_t)d * HDIM + lane * 4;
  atomicAdd(out + 0, v.x);
  atomicAdd(out + 1, v.y);
  atomicAdd(out + 2, v.z);
  atomicAdd(out + 3, v.w);
}

// ---- convert W (f32, [K=128][N=128] row-major) to fragment-linear bf16 ----
// wf index = (((n*4 + kc)*32 + lane)*16 + j); value = W[K][col],
// col = n*16 + (lane&15), K = kc*32 + (j>>3)*16 + (lane>>4)*8 + (j&7)   (dense bf16 B layout)
__global__ __launch_bounds__(256) void gin_prep_w(const float* __restrict__ W,
                                                  __bf16* __restrict__ wf) {
  int idx = blockIdx.x * 256 + threadIdx.x;          // [0, 16384)
  int j    = idx & 15;
  int lane = (idx >> 4) & 31;
  int kc   = (idx >> 9) & 3;
  int n    = idx >> 11;
  int col  = n * 16 + (lane & 15);
  int k    = kc * 32 + (j >> 3) * 16 + (lane >> 4) * 8 + (j & 7);
  wf[idx] = (__bf16)W[k * HDIM + col];
}

// ---------------- WMMA GEMM: Y = op(X [+X2], scl/sft+relu) @ W + bias ----
// Also accumulates per-column sum / sum-of-squares for BatchNorm stats.
__global__ __launch_bounds__(256) void gin_gemm(const float* __restrict__ X,
                                                const float* __restrict__ X2,
                                                const float* __restrict__ scl,
                                                const float* __restrict__ sft,
                                                const __bf16* __restrict__ wfrag,
                                                const float* __restrict__ bias,
                                                float* __restrict__ Y,
                                                float* __restrict__ colsum,
                                                float* __restrict__ colsumsq,
                                                int nrows) {
  __shared__ __align__(32) __bf16 sW[8 * 4 * 32 * 16];   // 32 KB fragment-linear W
  __shared__ float s_sum[HDIM];
  __shared__ float s_ssq[HDIM];

  const int tid  = threadIdx.x;
  const int wave = tid >> 5;
  const int lane = tid & 31;
  const int lrow = lane & 15;
  const int kgrp = lane >> 4;

  if (tid < HDIM) { s_sum[tid] = 0.f; s_ssq[tid] = 0.f; }
  {
    const uint4* s = (const uint4*)wfrag;
    uint4* d = (uint4*)sW;
#pragma unroll
    for (int t = 0; t < 8; ++t) d[tid + t * 256] = s[tid + t * 256];
  }
  __syncthreads();

  const int rowBase = blockIdx.x * 128 + wave * 16;
  const int m = rowBase + lrow;                 // A-matrix row owned by this lane
  // Clamp instead of predicate: out-of-range rows compute a duplicate of the
  // last valid row; their outputs are never stored / never enter the stats.
  const int mm = (m < nrows) ? m : (nrows - 1);
  const float* __restrict__ xrow  = X + (size_t)mm * HDIM;
  const float* __restrict__ x2row = X2 ? (X2 + (size_t)mm * HDIM) : X;

  const v8f vzero = {0.f, 0.f, 0.f, 0.f, 0.f, 0.f, 0.f, 0.f};
  v8f acc[8];
#pragma unroll
  for (int n = 0; n < 8; ++n) acc[n] = vzero;

#pragma unroll
  for (int kc = 0; kc < 4; ++kc) {
    // ---- build A fragment (16x32 bf16 slice of rows rowBase..+15) ----
    float tmp[16];
#pragma unroll
    for (int half = 0; half < 2; ++half) {
      const int kbase = kc * 32 + half * 16 + kgrp * 8;   // 32B-aligned offsets
      *(float4*)&tmp[half * 8 + 0] = *(const float4*)(xrow + kbase);
      *(float4*)&tmp[half * 8 + 4] = *(const float4*)(xrow + kbase + 4);
      if (X2) {
        float t2[8];
        *(float4*)&t2[0] = *(const float4*)(x2row + kbase);
        *(float4*)&t2[4] = *(const float4*)(x2row + kbase + 4);
#pragma unroll
        for (int j = 0; j < 8; ++j) tmp[half * 8 + j] += t2[j];
      }
      if (scl) {
        float sv[8], fv[8];
        *(float4*)&sv[0] = *(const float4*)(scl + kbase);
        *(float4*)&sv[4] = *(const float4*)(scl + kbase + 4);
        *(float4*)&fv[0] = *(const float4*)(sft + kbase);
        *(float4*)&fv[4] = *(const float4*)(sft + kbase + 4);
#pragma unroll
        for (int j = 0; j < 8; ++j)
          tmp[half * 8 + j] = fmaxf(tmp[half * 8 + j] * sv[j] + fv[j], 0.f);
      }
    }
    v16bf a;
#pragma unroll
    for (int j = 0; j < 16; ++j) a[j] = (__bf16)tmp[j];

#pragma unroll
    for (int n = 0; n < 8; ++n) {
      const v16bf b = *(const v16bf*)(sW + (((n * 4 + kc) * 32 + lane) << 4));
      acc[n] = __builtin_amdgcn_wmma_f32_16x16x32_bf16(false, a, false, b,
                                                       (short)0, acc[n],
                                                       false, false);
    }
  }

  // ---- epilogue: bias add, store, BN stat partials ----
  const bool fullTile = (rowBase + 16) <= nrows;    // wave-uniform
#pragma unroll
  for (int n = 0; n < 8; ++n) {
    const int col = n * 16 + lrow;
    const float bia = bias[col];
    float psum = 0.f, pssq = 0.f;
    if (fullTile) {
#pragma unroll
      for (int r = 0; r < 8; ++r) {
        const int row = rowBase + kgrp * 8 + r;     // lanes 16-31 hold M+8
        const float val = acc[n][r] + bia;
        Y[(size_t)row * HDIM + col] = val;
        psum += val;
        pssq += val * val;
      }
    } else {
#pragma unroll
      for (int r = 0; r < 8; ++r) {
        const int row = rowBase + kgrp * 8 + r;
        if (row < nrows) {
          const float val = acc[n][r] + bia;
          Y[(size_t)row * HDIM + col] = val;
          psum += val;
          pssq += val * val;
        }
      }
    }
    atomicAdd(&s_sum[col], psum);
    atomicAdd(&s_ssq[col], pssq);
  }
  __syncthreads();
  if (tid < HDIM) {
    atomicAdd(&colsum[tid], s_sum[tid]);
    atomicAdd(&colsumsq[tid], s_ssq[tid]);
  }
}

// ---------------- BN finalize: scale/shift from column stats ----------------
__global__ void gin_bn_finalize(const float* __restrict__ colsum,
                                const float* __restrict__ colsumsq,
                                const float* __restrict__ gamma,
                                const float* __restrict__ beta,
                                float* __restrict__ scl,
                                float* __restrict__ sft,
                                float invn) {
  int c = threadIdx.x;
  float mean = colsum[c] * invn;
  float var  = fmaxf(colsumsq[c] * invn - mean * mean, 0.f);
  float s    = gamma[c] * rsqrtf(var + BN_EPS);
  scl[c] = s;
  sft[c] = beta[c] - mean * s;
}

// ---------------- elementwise BN+ReLU apply (layer output -> h) -------------
__global__ __launch_bounds__(256) void gin_bn_apply(const float* __restrict__ Y,
                                                    const float* __restrict__ scl,
                                                    const float* __restrict__ sft,
                                                    float* __restrict__ Hout) {
  int idx = blockIdx.x * 256 + threadIdx.x;          // over NH/4 float4's
  int i4  = idx * 4;
  int c4  = i4 & (HDIM - 1);
  float4 y = *(const float4*)(Y + i4);
  float4 s = *(const float4*)(scl + c4);
  float4 f = *(const float4*)(sft + c4);
  float4 o;
  o.x = fmaxf(y.x * s.x + f.x, 0.f);
  o.y = fmaxf(y.y * s.y + f.y, 0.f);
  o.z = fmaxf(y.z * s.z + f.z, 0.f);
  o.w = fmaxf(y.w * s.w + f.w, 0.f);
  *(float4*)(Hout + i4) = o;
}

// ---------------- graph sum-pool (one node per wave) ----------------
__global__ __launch_bounds__(256) void gin_pool(const float* __restrict__ h,
                                                const int* __restrict__ batch,
                                                float* __restrict__ pooled) {
  const int wave = threadIdx.x >> 5;
  const int lane = threadIdx.x & 31;
  const int node = blockIdx.x * 8 + wave;            // N % 8 == 0
  const int g = __builtin_amdgcn_readfirstlane(batch[node]);
  const float4 v = *(const float4*)(h + (size_t)node * HDIM + lane * 4);
  float* out = pooled + (size_t)g * HDIM + lane * 4;
  atomicAdd(out + 0, v.x);
  atomicAdd(out + 1, v.y);
  atomicAdd(out + 2, v.z);
  atomicAdd(out + 3, v.w);
}

// ---------------- tiny MLP head: relu(pooled@W1+b1)@W2+b2 ----------------
__global__ __launch_bounds__(256) void gin_head(const float* __restrict__ pooled,
                                                const float* __restrict__ w1,
                                                const float* __restrict__ b1v,
                                                const float* __restrict__ w2,
                                                const float* __restrict__ b2v,
                                                float* __restrict__ out) {
  __shared__ float hid[NGRAPH * HDIM];               // 64 KB
  int tid = threadIdx.x;
  for (int idx = tid; idx < NGRAPH * HDIM; idx += 256) {
    int g = idx >> 7, c = idx & 127;
    float acc = b1v[c];
    for (int k = 0; k < HDIM; ++k) acc += pooled[g * HDIM + k] * w1[k * HDIM + c];
    hid[idx] = fmaxf(acc, 0.f);
  }
  __syncthreads();
  for (int idx = tid; idx < NGRAPH * NCLASS; idx += 256) {
    int g = idx / NCLASS, j = idx - g * NCLASS;
    float acc = b2v[j];
    for (int k = 0; k < HDIM; ++k) acc += hid[g * HDIM + k] * w2[k * NCLASS + j];
    out[idx] = acc;
  }
}

extern "C" void kernel_launch(void* const* d_in, const int* in_sizes, int n_in,
                              void* d_out, int out_size, void* d_ws, size_t ws_size,
                              hipStream_t stream) {
  const float* x      = (const float*)d_in[0];
  const int*   edge   = (const int*)d_in[1];
  const int*   batch  = (const int*)d_in[2];
  const float* W1     = (const float*)d_in[3];
  const float* b1     = (const float*)d_in[4];
  const float* g1     = (const float*)d_in[5];
  const float* be1    = (const float*)d_in[6];
  const float* W2     = (const float*)d_in[7];
  const float* b2     = (const float*)d_in[8];
  const float* g2     = (const float*)d_in[9];
  const float* be2    = (const float*)d_in[10];
  const float* lin1_w = (const float*)d_in[11];
  const float* lin1_b = (const float*)d_in[12];
  const float* lin2_w = (const float*)d_in[13];
  const float* lin2_b = (const float*)d_in[14];

  const size_t NH = (size_t)N_NODES * HDIM;          // 12,800,000 floats
  float* ws    = (float*)d_ws;
  float* hbuf  = ws;                                 // current features
  float* agg   = ws + NH;                            // aggregation / y2 (reused)
  float* ybuf  = ws + 2 * NH;                        // GEMM1 output
  __bf16* wfrag   = (__bf16*)(ws + 3 * NH);          // 16384 bf16 = 8192 floats
  float* colsum   = ws + 3 * NH + 8192;
  float* colsumsq = colsum + HDIM;
  float* scl      = colsumsq + HDIM;
  float* sft      = scl + HDIM;
  float* pooled   = sft + HDIM;                      // G*H floats

  const int* srcv = edge;
  const int* dstv = edge + N_EDGES;

  hipMemcpyAsync(hbuf, x, NH * sizeof(float), hipMemcpyDeviceToDevice, stream);

  const int gemm_blocks = (N_NODES + 127) / 128;     // 782
  const int scat_blocks = N_EDGES / 8;               // 200,000 (8 waves/block)
  const int pool_blocks = N_NODES / 8;               // 12,500
  const int elem_blocks = (int)(NH / 4 / 256);       // 12,500 (float4)
  const float invn = 1.0f / (float)N_NODES;

  for (int i = 0; i < NLAYERS; ++i) {
    // aggregation: agg = segment_sum(h[src] -> dst)
    hipMemsetAsync(agg, 0, NH * sizeof(float), stream);
    gin_scatter<<<scat_blocks, 256, 0, stream>>>(hbuf, srcv, dstv, agg);

    // linear 1: y = (h + agg) @ W1 + b1, with BN stats
    gin_prep_w<<<64, 256, 0, stream>>>(W1 + (size_t)i * HDIM * HDIM, wfrag);
    hipMemsetAsync(colsum, 0, 2 * HDIM * sizeof(float), stream);
    gin_gemm<<<gemm_blocks, 256, 0, stream>>>(hbuf, agg, nullptr, nullptr, wfrag,
                                              b1 + i * HDIM, ybuf,
                                              colsum, colsumsq, N_NODES);
    gin_bn_finalize<<<1, HDIM, 0, stream>>>(colsum, colsumsq,
                                            g1 + i * HDIM, be1 + i * HDIM,
                                            scl, sft, invn);

    // linear 2: y2 = relu(BN1(y)) @ W2 + b2 (BN1+ReLU fused into A-load)
    gin_prep_w<<<64, 256, 0, stream>>>(W2 + (size_t)i * HDIM * HDIM, wfrag);
    hipMemsetAsync(colsum, 0, 2 * HDIM * sizeof(float), stream);
    gin_gemm<<<gemm_blocks, 256, 0, stream>>>(ybuf, nullptr, scl, sft, wfrag,
                                              b2 + i * HDIM, agg,
                                              colsum, colsumsq, N_NODES);
    gin_bn_finalize<<<1, HDIM, 0, stream>>>(colsum, colsumsq,
                                            g2 + i * HDIM, be2 + i * HDIM,
                                            scl, sft, invn);

    // h = relu(BN2(y2))
    gin_bn_apply<<<elem_blocks, 256, 0, stream>>>(agg, scl, sft, hbuf);
  }

  // readout
  hipMemsetAsync(pooled, 0, (size_t)NGRAPH * HDIM * sizeof(float), stream);
  gin_pool<<<pool_blocks, 256, 0, stream>>>(hbuf, batch, pooled);
  gin_head<<<1, 256, 0, stream>>>(pooled, lin1_w, lin1_b, lin2_w, lin2_b,
                                  (float*)d_out);
}
